// LengthRegulator_4672924418202
// MI455X (gfx1250) — matrix operand order repaired and checked by reading, not compiled, not run on hardware
//
#include <hip/hip_runtime.h>
#include <hip/hip_bf16.h>
#include <cstdint>

typedef __attribute__((ext_vector_type(16))) _Float16 v16h;
typedef __attribute__((ext_vector_type(8)))  _Float16 v8h;
typedef __attribute__((ext_vector_type(8)))  float    v8f;

#define BATCH   64
#define SEQ     1024
#define DCH     384          // channels (D == F_SZ)
#define KDIM    1152         // 3 * 384
#define NCHUNK  36           // 1152 / 32
#define MROWS   (BATCH*SEQ)  // 65536
#define MAXOUT  3072
#define OUT_ELEMS (BATCH*(size_t)MAXOUT*DCH)   // 75497472
#define DUR_ELEMS (BATCH*SEQ)                  // 65536

// ---------------- pack x * mask -> f16 ----------------
__global__ __launch_bounds__(256)
void dp_pack_x(const float* __restrict__ x, const float* __restrict__ mask,
               _Float16* __restrict__ xh)
{
    size_t e = ((size_t)blockIdx.x * blockDim.x + threadIdx.x) * 4;
    if (e >= (size_t)MROWS * DCH) return;
    size_t p = e / DCH;
    float m = mask[p];
    float4 v = *(const float4*)(x + e);
    xh[e+0] = (_Float16)(v.x * m);
    xh[e+1] = (_Float16)(v.y * m);
    xh[e+2] = (_Float16)(v.z * m);
    xh[e+3] = (_Float16)(v.w * m);
}

// ---- pack weights (F, D, K) -> Wt[f][kw*384 + d] f16 ----
__global__ __launch_bounds__(256)
void dp_pack_w(const float* __restrict__ w, _Float16* __restrict__ wt)
{
    int gid = blockIdx.x * blockDim.x + threadIdx.x;
    if (gid >= DCH * KDIM) return;
    int f   = gid / KDIM;
    int rem = gid % KDIM;
    int kw  = rem / DCH;
    int d   = rem % DCH;
    wt[gid] = (_Float16)w[(size_t)f * KDIM + d * 3 + kw];
}

// Stage one 4KB B-chunk (64 cols x 32 k, f16) into LDS asynchronously.
// Layout in LDS: [j=0..3][lane=0..31][16 halves] so each lane's fragment read
// is one contiguous 32-byte access. 256 threads x 16B = 4KB exactly.
__device__ __forceinline__ void dp_stage_B(const _Float16* __restrict__ Wt,
                                           int f0, int c,
                                           _Float16* buf, int t)
{
    const int j    = t >> 6;
    const int ln   = (t >> 1) & 31;
    const int part = t & 1;
    const int half = ln >> 4;
    const int lm   = ln & 15;
    const _Float16* src = Wt + (size_t)(f0 + j * 16 + lm) * KDIM
                             + c * 32 + half * 8 + part * 16;
    unsigned lds = (unsigned)(size_t)(buf + ((j * 32 + ln) * 16 + part * 8));
    asm volatile("global_load_async_to_lds_b128 %0, %1, off"
                 :: "v"(lds), "v"(src) : "memory");
}

// ---------------- conv-as-GEMM via WMMA ----------------
// Y[p][f] = sum_{kw,d} X[p+kw-1][d] * Wt[f][kw*384+d]   (zero pad at seq edges)
// Block = 8 waves = 128 rows x 64 cols; B chunks staged via async-LDS,
// double buffered; each wave does 4 WMMAs per 32-wide K chunk (36 chunks).
__global__ __launch_bounds__(256)
void dp_conv_gemm_wmma(const _Float16* __restrict__ X,
                       const _Float16* __restrict__ Wt,
                       float* __restrict__ Y)
{
    __shared__ _Float16 Bsh[2][2048];       // 2 x 4KB double buffer

    const int t    = threadIdx.x;
    const int wv   = t >> 5;
    const int lane = t & 31;
    const int half = lane >> 4;
    const int lm   = lane & 15;

    const int bt = blockIdx.x / 6;          // 128-row block index
    const int cg = blockIdx.x % 6;          // 64-col group
    const int p0 = bt * 128 + wv * 16;      // this wave's 16-row tile
    const int f0 = cg << 6;

    const int r = p0 + lm;                  // A-matrix row (global position)
    const int s = r & (SEQ - 1);

    v8f acc0 = {}; v8f acc1 = {}; v8f acc2 = {}; v8f acc3 = {};

    dp_stage_B(Wt, f0, 0, Bsh[0], t);       // prologue: chunk 0

    #pragma unroll
    for (int kw = 0; kw < 3; ++kw) {
        const int  sp = s + kw - 1;
        const bool va = (unsigned)sp < (unsigned)SEQ;
        const _Float16* arow = X + (size_t)(r + kw - 1) * DCH;
        #pragma unroll
        for (int dc = 0; dc < 12; ++dc) {
            const int c = kw * 12 + dc;

            // my async writes for chunk c done -> barrier makes all visible,
            // and previous reads of the other buffer are complete.
            asm volatile("s_wait_asynccnt 0" ::: "memory");
            __syncthreads();

            if (c + 1 < NCHUNK)
                dp_stage_B(Wt, f0, c + 1, Bsh[(c + 1) & 1], t);

            v16h a;
            if (va) {
                const _Float16* srcp = arow + dc * 32 + half * 8;
                if (dc + 1 < 12) __builtin_prefetch(srcp + 32, 0, 1);
                v8h lo = *(const v8h*)(srcp);        // K = base+{0..7}/{8..15}
                v8h hi = *(const v8h*)(srcp + 16);   // K = base+{16..23}/{24..31}
                #pragma unroll
                for (int i = 0; i < 8; ++i) { a[i] = lo[i]; a[8 + i] = hi[i]; }
            } else {
                #pragma unroll
                for (int i = 0; i < 16; ++i) a[i] = (_Float16)0.0f;
            }

            const _Float16* bb = &Bsh[c & 1][lane * 16];
            v16h b0 = *(const v16h*)(bb);
            v16h b1 = *(const v16h*)(bb + 32 * 16);
            v16h b2 = *(const v16h*)(bb + 64 * 16);
            v16h b3 = *(const v16h*)(bb + 96 * 16);
            acc0 = __builtin_amdgcn_wmma_f32_16x16x32_f16(false, a, false, b0, (short)0, acc0, false, false);
            acc1 = __builtin_amdgcn_wmma_f32_16x16x32_f16(false, a, false, b1, (short)0, acc1, false, false);
            acc2 = __builtin_amdgcn_wmma_f32_16x16x32_f16(false, a, false, b2, (short)0, acc2, false, false);
            acc3 = __builtin_amdgcn_wmma_f32_16x16x32_f16(false, a, false, b3, (short)0, acc3, false, false);
        }
    }

    #pragma unroll
    for (int i = 0; i < 8; ++i) {
        int row = p0 + half * 8 + i;               // C/D layout: M = half*8 + i
        float* yrow = Y + (size_t)row * DCH + f0 + lm;
        yrow[0]  = acc0[i];
        yrow[16] = acc1[i];
        yrow[32] = acc2[i];
        yrow[48] = acc3[i];
    }
}

// ---------------- bias + relu + layernorm -> f16 ----------------
__global__ __launch_bounds__(256)
void dp_bias_relu_ln(const float* __restrict__ Y, const float* __restrict__ bias,
                     const float* __restrict__ g, const float* __restrict__ bl,
                     _Float16* __restrict__ H)
{
    const int row  = blockIdx.x * 8 + (threadIdx.x >> 5);
    const int lane = threadIdx.x & 31;
    const float* y = Y + (size_t)row * DCH;
    float v[12];
    float sum = 0.f, sq = 0.f;
    #pragma unroll
    for (int j = 0; j < 12; ++j) {
        int f = lane + 32 * j;
        float t = fmaxf(y[f] + bias[f], 0.f);
        v[j] = t; sum += t; sq += t * t;
    }
    #pragma unroll
    for (int o = 16; o > 0; o >>= 1) {
        sum += __shfl_xor(sum, o, 32);
        sq  += __shfl_xor(sq,  o, 32);
    }
    const float mean = sum * (1.0f / DCH);
    const float var  = sq * (1.0f / DCH) - mean * mean;
    const float inv  = rsqrtf(var + 1e-5f);
    _Float16* h = H + (size_t)row * DCH;
    #pragma unroll
    for (int j = 0; j < 12; ++j) {
        int f = lane + 32 * j;
        h[f] = (_Float16)((v[j] - mean) * inv * g[f] + bl[f]);
    }
}

// ---------------- final projection + duration ----------------
__global__ __launch_bounds__(256)
void dp_lin_dur(const _Float16* __restrict__ H, const float* __restrict__ lw,
                const float* __restrict__ lb, const float* __restrict__ mask,
                float* __restrict__ dur_out, int* __restrict__ reps)
{
    const int row  = blockIdx.x * 8 + (threadIdx.x >> 5);
    const int lane = threadIdx.x & 31;
    const _Float16* h = H + (size_t)row * DCH;
    float acc = 0.f;
    #pragma unroll
    for (int j = 0; j < 12; ++j) {
        int f = lane + 32 * j;
        acc += (float)h[f] * lw[f];
    }
    #pragma unroll
    for (int o = 16; o > 0; o >>= 1) acc += __shfl_xor(acc, o, 32);
    if (lane == 0) {
        float d   = (acc + lb[0]) * mask[row];
        float dur = fmaxf(expf(d) - 1.0f, 0.0f);
        dur_out[row] = dur;
        reps[row] = (int)rintf(dur);   // round-half-even like jnp.round
    }
}

// ---------------- per-batch inclusive scan ----------------
__global__ void dp_cumsum(const int* __restrict__ reps, int* __restrict__ cum,
                          int* __restrict__ tot)
{
    __shared__ int sh[SEQ];
    const int b = blockIdx.x, t = threadIdx.x;
    sh[t] = reps[b * SEQ + t];
    __syncthreads();
    for (int off = 1; off < SEQ; off <<= 1) {
        int v = (t >= off) ? sh[t - off] : 0;
        __syncthreads();
        sh[t] += v;
        __syncthreads();
    }
    cum[b * SEQ + t] = sh[t];
    if (t == SEQ - 1) tot[b] = sh[t];
}

// ---------------- searchsorted + output_pos ----------------
__global__ __launch_bounds__(256)
void dp_expand_idx(const int* __restrict__ cum, const int* __restrict__ tot,
                   int* __restrict__ idx_out, int* __restrict__ pos_out, int wide)
{
    int gid = blockIdx.x * blockDim.x + threadIdx.x;
    if (gid >= BATCH * MAXOUT) return;
    int b = gid / MAXOUT, t = gid % MAXOUT;
    const int* c = cum + b * SEQ;
    int lo = 0, hi = SEQ;
    while (lo < hi) { int mid = (lo + hi) >> 1; if (c[mid] <= t) lo = mid + 1; else hi = mid; }
    int idx = lo < (SEQ - 1) ? lo : (SEQ - 1);
    idx_out[gid] = idx;
    int pos = (t < tot[b]) ? (t + 1) : 0;
    if (wide) { pos_out[2 * gid] = pos; pos_out[2 * gid + 1] = 0; }
    else      { pos_out[gid] = pos; }
}

// ---------------- gather rows of x ----------------
__global__ __launch_bounds__(128)
void dp_gather(const float* __restrict__ x, const int* __restrict__ idx,
               const int* __restrict__ tot, float* __restrict__ out)
{
    const int row = blockIdx.x;           // b*MAXOUT + t
    const int b = row / MAXOUT, t = row % MAXOUT;
    const float scale = (t < tot[b]) ? 1.0f : 0.0f;
    const float4* src = (const float4*)(x + ((size_t)b * SEQ + idx[row]) * DCH);
    float4* dst = (float4*)(out + (size_t)row * DCH);
    for (int i = threadIdx.x; i < DCH / 4; i += blockDim.x) {
        float4 v = src[i];
        v.x *= scale; v.y *= scale; v.z *= scale; v.w *= scale;
        dst[i] = v;
    }
}

extern "C" void kernel_launch(void* const* d_in, const int* in_sizes, int n_in,
                              void* d_out, int out_size, void* d_ws, size_t ws_size,
                              hipStream_t stream) {
    const float* x    = (const float*)d_in[0];
    const float* mask = (const float*)d_in[1];
    const float* w1   = (const float*)d_in[2];
    const float* b1   = (const float*)d_in[3];
    const float* g1   = (const float*)d_in[4];
    const float* bb1  = (const float*)d_in[5];
    const float* w2   = (const float*)d_in[6];
    const float* b2   = (const float*)d_in[7];
    const float* g2   = (const float*)d_in[8];
    const float* bb2  = (const float*)d_in[9];
    const float* lw   = (const float*)d_in[10];
    const float* lb   = (const float*)d_in[11];

    char* ws = (char*)d_ws;
    size_t off = 0;
    auto take = [&](size_t bytes) { size_t o = off; off = (off + bytes + 255) & ~(size_t)255; return o; };
    const size_t F16ACT = (size_t)MROWS * DCH * sizeof(_Float16);   // 48 MB
    _Float16* XH  = (_Float16*)(ws + take(F16ACT));   // also reused as H2
    _Float16* H1  = (_Float16*)(ws + take(F16ACT));
    float*    Yb  = (float*)   (ws + take((size_t)MROWS * DCH * sizeof(float)));
    _Float16* W1t = (_Float16*)(ws + take((size_t)DCH * KDIM * sizeof(_Float16)));
    _Float16* W2t = (_Float16*)(ws + take((size_t)DCH * KDIM * sizeof(_Float16)));
    int*      REP = (int*)     (ws + take((size_t)MROWS * sizeof(int)));
    int*      CUM = (int*)     (ws + take((size_t)MROWS * sizeof(int)));
    int*      TOT = (int*)     (ws + take((size_t)BATCH * sizeof(int)));
    int*      IDX = (int*)     (ws + take((size_t)BATCH * MAXOUT * sizeof(int)));
    _Float16* H2  = XH;   // xh dead after first GEMM

    // output tuple layout: (output f32, output_pos int32|int64, duration f32)
    long long posSlots = (long long)out_size - (long long)OUT_ELEMS - (long long)DUR_ELEMS;
    int wide = (posSlots >= 2LL * BATCH * MAXOUT) ? 1 : 0;
    if (posSlots < (long long)BATCH * MAXOUT) posSlots = (long long)BATCH * MAXOUT;
    float* out_main = (float*)d_out;
    int*   out_pos  = (int*)(out_main + OUT_ELEMS);
    float* out_dur  = out_main + OUT_ELEMS + (size_t)posSlots;

    // 1) pack inputs / weights to f16
    dp_pack_x<<<((size_t)MROWS * DCH / 4 + 255) / 256, 256, 0, stream>>>(x, mask, XH);
    dp_pack_w<<<(DCH * KDIM + 255) / 256, 256, 0, stream>>>(w1, W1t);
    dp_pack_w<<<(DCH * KDIM + 255) / 256, 256, 0, stream>>>(w2, W2t);

    // 2) conv1 (WMMA GEMM, async-LDS staged B) + epilogue
    const int gemmBlocks = (MROWS / 128) * 6;   // 512 row blocks x 6 col groups
    dp_conv_gemm_wmma<<<gemmBlocks, 256, 0, stream>>>(XH, W1t, Yb);
    dp_bias_relu_ln<<<MROWS / 8, 256, 0, stream>>>(Yb, b1, g1, bb1, H1);

    // 3) conv2 (WMMA GEMM) + epilogue  (H2 aliases XH)
    dp_conv_gemm_wmma<<<gemmBlocks, 256, 0, stream>>>(H1, W2t, Yb);
    dp_bias_relu_ln<<<MROWS / 8, 256, 0, stream>>>(Yb, b2, g2, bb2, H2);

    // 4) projection -> duration, repeats
    dp_lin_dur<<<MROWS / 8, 256, 0, stream>>>(H2, lw, lb, mask, out_dur, REP);

    // 5) cumsum per batch
    dp_cumsum<<<BATCH, SEQ, 0, stream>>>(REP, CUM, TOT);

    // 6) searchsorted + output_pos
    dp_expand_idx<<<(BATCH * MAXOUT + 255) / 256, 256, 0, stream>>>(CUM, TOT, IDX, out_pos, wide);

    // 7) gather output rows
    dp_gather<<<BATCH * MAXOUT, 128, 0, stream>>>(x, IDX, TOT, out_main);

    (void)n_in; (void)in_sizes; (void)ws_size;
}
